// decoder_22617297781038
// MI455X (gfx1250) — compile-verified, matrix operand order checked by reading, not run on hardware
//
#include <hip/hip_runtime.h>
#include <hip/hip_bf16.h>

// ---------------------------------------------------------------------------
// Problem constants (from the reference)
// ---------------------------------------------------------------------------
namespace {
constexpr int kB     = 128;
constexpr int kTV    = 64;
constexpr int kTA    = 128;
constexpr int kEV    = 1024;
constexpr int kEA    = 512;
constexpr int kH     = 1024;
constexpr int kEMB   = 512;
constexpr int kC     = 512;
constexpr int kIN    = kEMB + kC;   // 1024
constexpr int kVOCAB = 20000;
}

// ---------------------------------------------------------------------------
// WMMA / vector types (CDNA5 / gfx1250, wave32)
// ---------------------------------------------------------------------------
typedef __bf16          v16bf  __attribute__((ext_vector_type(16)));
typedef float           v8f    __attribute__((ext_vector_type(8)));
typedef unsigned short  u16x4  __attribute__((ext_vector_type(4)));
typedef unsigned short  u16x8  __attribute__((ext_vector_type(8)));
typedef unsigned short  u16x16 __attribute__((ext_vector_type(16)));
typedef unsigned int    u32x4  __attribute__((ext_vector_type(4)));
typedef unsigned int    u32x8  __attribute__((ext_vector_type(8)));

// fp32 -> bf16 (round-to-nearest-even), kept as raw u16 so LDS traffic is typeless
__device__ __forceinline__ unsigned short f2bfu(float x) {
    union { float f; unsigned u; } c; c.f = x;
    unsigned u = c.u;
    unsigned r = u + 0x7fffu + ((u >> 16) & 1u);
    return (unsigned short)(r >> 16);
}

__device__ __forceinline__ void store_bf4(unsigned short* dst, float4 v) {
    u16x4 p;
    p.x = f2bfu(v.x); p.y = f2bfu(v.y); p.z = f2bfu(v.z); p.w = f2bfu(v.w);
    *reinterpret_cast<u16x4*>(dst) = p;
}

// ---------------------------------------------------------------------------
// Generic GEMM: D[M,N] = A[M,K] * B[N,K]^T (+ bias[N]) (+= D if accumulate)
// A, B fp32 row-major with ld == K; converted to bf16 on the LDS stage.
// Block: 128 threads (4 waves), 64x64 output tile, each wave does 32x32 via
// 2x2 V_WMMA_F32_16X16X32_BF16 accumulators.
//
// Fragment gathers follow the CDNA5 16-bit layouts (ISA 7.12.2):
//   A 16x32 : lane<16 -> M=lane,   K = {0..7, 16..23}; lane>=16 -> K = {8..15, 24..31}
//   B 32x16 : lane<16 -> N=lane, same K halves (B loaded column-major == rows of W)
// ---------------------------------------------------------------------------
#define LDSK 40   // 32 K-elems + 8 pad (keeps 16B alignment, spreads banks)

__global__ __launch_bounds__(128)
void wmma_gemm_bf16(const float* __restrict__ A, const float* __restrict__ Bm,
                    const float* __restrict__ bias, float* __restrict__ D,
                    int M, int N, int K, int ldd, int accumulate)
{
    __shared__ __align__(16) unsigned short As[64 * LDSK];
    __shared__ __align__(16) unsigned short Bs[64 * LDSK];

    const int tid    = threadIdx.x;
    const int wave   = tid >> 5;
    const int lane   = tid & 31;
    const int waveM  = (wave >> 1) * 32;       // 0 or 32
    const int waveN  = (wave & 1) * 32;        // 0 or 32
    const int lrow   = lane & 15;              // row within 16x16 subtile
    const int khalf  = (lane >> 4) * 8;        // 0 or 8 -> K halves

    const int blockM = blockIdx.y * 64;
    const int blockN = blockIdx.x * 64;

    // LDS fill mapping: 128 threads, each loads float4; 16 rows per pass
    const int frow = tid >> 3;                 // 0..15
    const int fcol = (tid & 7) * 4;            // 0,4,...,28

    // block-uniform interior test: every M here is a multiple of 64; only the
    // vocab projection has a ragged N edge, so nearly all blocks take the
    // unguarded path (pure b128 loads feeding the WMMA pipeline).
    const bool interior = (blockM + 64 <= M) && (blockN + 64 <= N);

    const float* aBase = A  + (size_t)(blockM + frow) * K + fcol;
    const float* bBase = Bm + (size_t)(blockN + frow) * K + fcol;

    v8f acc[2][2];
    #pragma unroll
    for (int mi = 0; mi < 2; ++mi)
        #pragma unroll
        for (int ni = 0; ni < 2; ++ni)
            acc[mi][ni] = (v8f){0.f, 0.f, 0.f, 0.f, 0.f, 0.f, 0.f, 0.f};

    for (int kt = 0; kt < K; kt += 32) {
        __syncthreads();   // protect LDS against previous iteration's readers

        // ---- stage A tile (64 x 32) and B tile (64 x 32), fp32 -> bf16 ----
        if (interior) {
            #pragma unroll
            for (int rr = 0; rr < 64; rr += 16) {
                const float* ap = aBase + (size_t)rr * K + kt;
                const float* bp = bBase + (size_t)rr * K + kt;
                float4 va = *reinterpret_cast<const float4*>(ap);
                float4 vb = *reinterpret_cast<const float4*>(bp);
                if (kt + 32 < K) {
                    __builtin_prefetch(ap + 32, 0, 1);
                    __builtin_prefetch(bp + 32, 0, 1);
                }
                store_bf4(&As[(frow + rr) * LDSK + fcol], va);
                store_bf4(&Bs[(frow + rr) * LDSK + fcol], vb);
            }
        } else {
            #pragma unroll
            for (int rr = 0; rr < 64; rr += 16) {
                int r  = frow + rr;
                int gm = blockM + r;
                float4 va = make_float4(0.f, 0.f, 0.f, 0.f);
                if (gm < M) va = *reinterpret_cast<const float4*>(A + (size_t)gm * K + kt + fcol);
                store_bf4(&As[r * LDSK + fcol], va);

                int gn = blockN + r;
                float4 vb = make_float4(0.f, 0.f, 0.f, 0.f);
                if (gn < N) vb = *reinterpret_cast<const float4*>(Bm + (size_t)gn * K + kt + fcol);
                store_bf4(&Bs[r * LDSK + fcol], vb);
            }
        }
        __syncthreads();

        // ---- gather fragments and issue 4 WMMAs ----
        v16bf aF[2], bF[2];
        #pragma unroll
        for (int mi = 0; mi < 2; ++mi) {
            const unsigned short* p = &As[(waveM + mi * 16 + lrow) * LDSK];
            u16x8 lo = *reinterpret_cast<const u16x8*>(p + khalf);
            u16x8 hi = *reinterpret_cast<const u16x8*>(p + khalf + 16);
            u16x16 cat = __builtin_shufflevector(lo, hi, 0,1,2,3,4,5,6,7,8,9,10,11,12,13,14,15);
            aF[mi] = __builtin_bit_cast(v16bf, cat);
        }
        #pragma unroll
        for (int ni = 0; ni < 2; ++ni) {
            const unsigned short* p = &Bs[(waveN + ni * 16 + lrow) * LDSK];
            u16x8 lo = *reinterpret_cast<const u16x8*>(p + khalf);
            u16x8 hi = *reinterpret_cast<const u16x8*>(p + khalf + 16);
            u16x16 cat = __builtin_shufflevector(lo, hi, 0,1,2,3,4,5,6,7,8,9,10,11,12,13,14,15);
            bF[ni] = __builtin_bit_cast(v16bf, cat);
        }

        #pragma unroll
        for (int mi = 0; mi < 2; ++mi)
            #pragma unroll
            for (int ni = 0; ni < 2; ++ni)
                acc[mi][ni] = __builtin_amdgcn_wmma_f32_16x16x32_bf16(
                    /*neg_a=*/false, aF[mi], /*neg_b=*/false, bF[ni],
                    /*c_mod=*/(short)0, acc[mi][ni],
                    /*reuse_a=*/false, /*reuse_b=*/false);
    }

    // ---- store: C/D layout — VGPR r, lane<16 -> M=r, lane>=16 -> M=8+r ----
    #pragma unroll
    for (int mi = 0; mi < 2; ++mi)
        #pragma unroll
        for (int ni = 0; ni < 2; ++ni) {
            int gm0 = blockM + waveM + mi * 16 + (lane >> 4) * 8;
            int gn  = blockN + waveN + ni * 16 + (lane & 15);
            #pragma unroll
            for (int r = 0; r < 8; ++r) {
                int gm = gm0 + r;
                if (gm < M && gn < N) {
                    float v = acc[mi][ni][r];
                    if (bias) v += bias[gn];
                    size_t off = (size_t)gm * ldd + gn;
                    if (accumulate) v += D[off];
                    D[off] = v;
                }
            }
        }
}

// ---------------------------------------------------------------------------
// Bahdanau attention tail: scores = W3 . tanh(V1[b] + V2[b,t]), softmax over T,
// ctx[b,e] = sum_t attn[t] * enc[b,t,e].   One block per batch item.
//
// The context stage pulls enc[b, 0:T, ec:ec+128] tiles into LDS with the
// Tensor Data Mover (tensor_load_to_lds, TENSORcnt), then reduces from LDS.
// ---------------------------------------------------------------------------
#define ECHUNK 128

__global__ __launch_bounds__(256)
void attend_kernel(const float* __restrict__ V1, const float* __restrict__ V2,
                   const float* __restrict__ W3, const float* __restrict__ enc,
                   float* __restrict__ ctx, int T, int E, int C)
{
    const int b    = blockIdx.x;
    const int tid  = threadIdx.x;
    const int wave = tid >> 5;
    const int lane = tid & 31;

    __shared__ float sS[128];     // T <= 128
    __shared__ float sRed[8];
    __shared__ __align__(16) float sEnc[128 * ECHUNK];   // 64 KB tile buffer

    const float* v1 = V1 + (size_t)b * C;
    for (int t = wave; t < T; t += 8) {
        const float* v2 = V2 + (size_t)(b * T + t) * C;
        float p = 0.f;
        for (int c = lane; c < C; c += 32)
            p += W3[c] * tanhf(v1[c] + v2[c]);
        #pragma unroll
        for (int off = 16; off; off >>= 1) p += __shfl_down(p, off, 32);
        if (lane == 0) sS[t] = p;
    }
    __syncthreads();

    // softmax over T (threads >= T contribute -inf / 0)
    float v = (tid < T) ? sS[tid] : -3.0e38f;
    float m = v;
    #pragma unroll
    for (int off = 16; off; off >>= 1) m = fmaxf(m, __shfl_down(m, off, 32));
    if (lane == 0) sRed[wave] = m;
    __syncthreads();
    float bm = sRed[0];
    #pragma unroll
    for (int w = 1; w < 8; ++w) bm = fmaxf(bm, sRed[w]);

    float e = (tid < T) ? __expf(v - bm) : 0.f;
    float s = e;
    #pragma unroll
    for (int off = 16; off; off >>= 1) s += __shfl_down(s, off, 32);
    __syncthreads();
    if (lane == 0) sRed[wave] = s;
    __syncthreads();
    float bs = 0.f;
    #pragma unroll
    for (int w = 0; w < 8; ++w) bs += sRed[w];

    if (tid < T) sS[tid] = e / bs;
    __syncthreads();

    // ---- weighted context via TDM tiles: enc[b, 0:T, ec:ec+ECHUNK] -> LDS ----
    for (int ec = 0; ec < E; ec += ECHUNK) {
        if (wave == 0) {
            // Tensor DMA descriptor (ISA 8.3/8.4): 2D tile, fp32 elements.
            unsigned long long ga = (unsigned long long)(uintptr_t)
                                    (enc + (size_t)b * T * E + ec);   // tile start
            unsigned lds = (unsigned)(uintptr_t)sEnc;                 // LDS byte offset

            u32x4 g0;
            g0[0] = 1u;                                   // count=1, user descriptor
            g0[1] = lds;                                  // lds_addr
            g0[2] = (unsigned)ga;                         // global_addr[31:0]
            g0[3] = (unsigned)((ga >> 32) & 0x01FFFFFFu)  // global_addr[56:32]
                    | 0x80000000u;                        // type=2 ("image")

            u32x8 g1;
            g1[0] = (2u << 16);                           // data_size=4B, no multicast
            g1[1] = ((unsigned)E & 0xFFFFu) << 16;        // tensor_dim0[15:0]
            g1[2] = ((unsigned)T & 0xFFFFu) << 16;        // dim0 hi=0 | tensor_dim1 lo
            g1[3] = ((unsigned)ECHUNK) << 16;             // dim1 hi=0 | tile_dim0
            g1[4] = (unsigned)T & 0xFFFFu;                // tile_dim1 | tile_dim2=0
            g1[5] = (unsigned)E;                          // tensor_dim0_stride[31:0]
            g1[6] = 0u;                                   // stride0 hi | stride1 lo
            g1[7] = 0u;

            u32x4 g2 = (u32x4){0u, 0u, 0u, 0u};           // dims 2/3 unused
            u32x4 g3 = (u32x4){0u, 0u, 0u, 0u};

            asm volatile("tensor_load_to_lds %0, %1, %2, %3"
                         :: "s"(g0), "s"(g1), "s"(g2), "s"(g3)
                         : "memory");
            __builtin_amdgcn_s_wait_tensorcnt(0);
        }
        __syncthreads();

        if (tid < ECHUNK) {
            float a = 0.f;
            for (int t = 0; t < T; ++t)
                a += sS[t] * sEnc[t * ECHUNK + tid];
            ctx[(size_t)b * E + ec + tid] = a;
        }
        __syncthreads();   // done with this LDS tile before the next DMA
    }
}

// ---------------------------------------------------------------------------
// Cross-modal attention over the 2 contexts + build RNN input
// x[b] = [ input[b,0,:EMB] , tanh(sum_j a_j * allctx[b,j,:]) ]
// ---------------------------------------------------------------------------
__global__ __launch_bounds__(128)
void cross_ctx_kernel(const float* __restrict__ CA1, const float* __restrict__ V2c,
                      const float* __restrict__ W3, const float* __restrict__ allctx,
                      const float* __restrict__ input, float* __restrict__ x,
                      int C, int EMB)
{
    const int b    = blockIdx.x;
    const int tid  = threadIdx.x;
    const int wave = tid >> 5;
    const int lane = tid & 31;
    __shared__ float sS[2];

    if (wave < 2) {
        const float* v1 = CA1 + (size_t)b * C;
        const float* v2 = V2c + (size_t)(b * 2 + wave) * C;
        float p = 0.f;
        for (int c = lane; c < C; c += 32)
            p += W3[c] * tanhf(v1[c] + v2[c]);
        #pragma unroll
        for (int off = 16; off; off >>= 1) p += __shfl_down(p, off, 32);
        if (lane == 0) sS[wave] = p;
    }
    __syncthreads();

    float m  = fmaxf(sS[0], sS[1]);
    float e0 = __expf(sS[0] - m), e1 = __expf(sS[1] - m);
    float a0 = e0 / (e0 + e1),   a1 = e1 / (e0 + e1);

    float* xb = x + (size_t)b * (EMB + C);
    for (int i = tid; i < EMB; i += 128)
        xb[i] = input[(size_t)b * EMB + i];
    const float* cv = allctx + (size_t)b * 2 * C;
    for (int c = tid; c < C; c += 128)
        xb[EMB + c] = tanhf(a0 * cv[c] + a1 * cv[C + c]);
}

// ---------------------------------------------------------------------------
// LSTM pointwise: gates already holds x@W_ih^T + b_ih + h0@W_hh^T + b_hh
// ---------------------------------------------------------------------------
__global__ __launch_bounds__(256)
void lstm_out_kernel(const float* __restrict__ gates, const float* __restrict__ c0,
                     float* __restrict__ h_new, float* __restrict__ c_new, int H)
{
    int idx = blockIdx.x * blockDim.x + threadIdx.x;   // over B*H
    int b = idx / H, h = idx % H;
    const float* g = gates + (size_t)b * 4 * H;
    float gi = g[h], gf = g[H + h], gg = g[2 * H + h], go = g[3 * H + h];
    float si = 1.f / (1.f + __expf(-gi));
    float sf = 1.f / (1.f + __expf(-gf));
    float so = 1.f / (1.f + __expf(-go));
    float cn = sf * c0[idx] + si * tanhf(gg);
    float hn = so * tanhf(cn);
    c_new[idx] = cn;
    h_new[idx] = hn;
}

// ---------------------------------------------------------------------------
// Launch
// ---------------------------------------------------------------------------
extern "C" void kernel_launch(void* const* d_in, const int* in_sizes, int n_in,
                              void* d_out, int out_size, void* d_ws, size_t ws_size,
                              hipStream_t stream)
{
    (void)in_sizes; (void)n_in; (void)out_size; (void)ws_size;

    const float* input  = (const float*)d_in[0];
    const float* encV   = (const float*)d_in[1];
    const float* encA   = (const float*)d_in[2];
    const float* h0     = (const float*)d_in[3];   // [1,B,H] contiguous
    const float* c0     = (const float*)d_in[4];
    const float* W_va1  = (const float*)d_in[5];
    const float* W_va2  = (const float*)d_in[6];
    const float* W_va3  = (const float*)d_in[7];
    const float* W_venc = (const float*)d_in[8];
    const float* W_aa1  = (const float*)d_in[9];
    const float* W_aa2  = (const float*)d_in[10];
    const float* W_aa3  = (const float*)d_in[11];
    const float* W_aenc = (const float*)d_in[12];
    const float* W_ca1  = (const float*)d_in[13];
    const float* W_ca2  = (const float*)d_in[14];
    const float* W_ca3  = (const float*)d_in[15];
    const float* W_ih   = (const float*)d_in[16];
    const float* W_hh   = (const float*)d_in[17];
    const float* b_ih   = (const float*)d_in[18];
    const float* b_hh   = (const float*)d_in[19];
    const float* W_out  = (const float*)d_in[20];
    const float* b_out  = (const float*)d_in[21];

    float* out    = (float*)d_out;
    float* logits = out;                                   // [B, VOCAB]
    float* h_new  = out + (size_t)kB * kVOCAB;             // [B, H]
    float* c_new  = h_new + (size_t)kB * kH;               // [B, H]

    float* ws     = (float*)d_ws;
    float* V1v    = ws; ws += (size_t)kB * kC;
    float* V1a    = ws; ws += (size_t)kB * kC;
    float* CA1    = ws; ws += (size_t)kB * kC;
    float* V2v    = ws; ws += (size_t)kB * kTV * kC;
    float* V2a    = ws; ws += (size_t)kB * kTA * kC;
    float* ctxv   = ws; ws += (size_t)kB * kEV;
    float* ctxa   = ws; ws += (size_t)kB * kEA;
    float* allctx = ws; ws += (size_t)kB * 2 * kC;
    float* V2c    = ws; ws += (size_t)kB * 2 * kC;
    float* xbuf   = ws; ws += (size_t)kB * kIN;
    float* gates  = ws; ws += (size_t)kB * 4 * kH;

    auto gemm = [&](const float* A, const float* Bm, const float* bias, float* D,
                    int M, int N, int K, int ldd, int acc) {
        dim3 grid((N + 63) / 64, (M + 63) / 64);
        wmma_gemm_bf16<<<grid, 128, 0, stream>>>(A, Bm, bias, D, M, N, K, ldd, acc);
    };

    // h0 projections for the three attentions + recurrent half of the gates
    gemm(h0, W_va1, nullptr, V1v, kB, kC, kH, kC, 0);
    gemm(h0, W_aa1, nullptr, V1a, kB, kC, kH, kC, 0);
    gemm(h0, W_ca1, nullptr, CA1, kB, kC, kH, kC, 0);
    gemm(h0, W_hh,  b_hh,    gates, kB, 4 * kH, kH, 4 * kH, 0);

    // encoder projections (the big GEMMs)
    gemm(encV, W_va2, nullptr, V2v, kB * kTV, kC, kEV, kC, 0);
    gemm(encA, W_aa2, nullptr, V2a, kB * kTA, kC, kEA, kC, 0);

    // attention tails -> raw contexts (TDM-fed)
    attend_kernel<<<kB, 256, 0, stream>>>(V1v, V2v, W_va3, encV, ctxv, kTV, kEV, kC);
    attend_kernel<<<kB, 256, 0, stream>>>(V1a, V2a, W_aa3, encA, ctxa, kTA, kEA, kC);

    // encoded contexts, interleaved into allctx[b, {0,1}, :] via ldd = 2C
    gemm(ctxv, W_venc, nullptr, allctx,      kB, kC, kEV, 2 * kC, 0);
    gemm(ctxa, W_aenc, nullptr, allctx + kC, kB, kC, kEA, 2 * kC, 0);

    // cross-modal attention: project both contexts (M = 2B rows), then fuse
    gemm(allctx, W_ca2, nullptr, V2c, kB * 2, kC, kC, kC, 0);
    cross_ctx_kernel<<<kB, 128, 0, stream>>>(CA1, V2c, W_ca3, allctx, input, xbuf, kC, kEMB);

    // input half of the gates, accumulated onto the recurrent half
    gemm(xbuf, W_ih, b_ih, gates, kB, 4 * kH, kIN, 4 * kH, 1);

    // LSTM pointwise -> h_new, c_new straight into d_out
    lstm_out_kernel<<<(kB * kH + 255) / 256, 256, 0, stream>>>(gates, c0, h_new, c_new, kH);

    // vocab projection
    gemm(h_new, W_out, b_out, logits, kB, kVOCAB, kH, kVOCAB, 0);
}
